// STAtt_Block_a_T_89893665505910
// MI455X (gfx1250) — matrix operand order repaired and checked by reading, not compiled, not run on hardware
//
#include <hip/hip_runtime.h>

// Problem constants (from reference)
#define NN 32
#define PP 6
#define CC 128
#define TT 64
#define VV 25
#define SS 3
#define CI 32
#define NP (NN * PP)            // 192
#define TV (TT * VV)            // 1600
#define TOT ((size_t)NP * CC * TV)  // 39,321,600 elements

typedef __bf16 bf16;
typedef bf16  v16bf __attribute__((ext_vector_type(16)));
typedef float v8f   __attribute__((ext_vector_type(8)));
typedef unsigned int u32x4 __attribute__((ext_vector_type(4)));
typedef int  i32x4 __attribute__((ext_vector_type(4)));
typedef int  i32x8 __attribute__((ext_vector_type(8)));

union Frag { v16bf v; int4 q[2]; };

#if defined(__has_builtin)
#if __has_builtin(__builtin_amdgcn_tensor_load_to_lds) && \
    __has_builtin(__builtin_amdgcn_s_wait_tensorcnt)
#define HAVE_TDM 1
#endif
#endif
#ifndef HAVE_TDM
#define HAVE_TDM 0
#endif

// ---------------------------------------------------------------------------
// Tensor Data Mover: async 2D tile load (global -> LDS), bf16 elements.
// D# group0: count=1 | lds_addr | global_addr | type=2
// D# group1: data_size=2B, tensor dims huge (no OOB clip), tile_x x tile_y,
//            row stride in elements.  Groups 2/3 zero (2D tensor).
// ---------------------------------------------------------------------------
__device__ inline void tdm_load_2d(unsigned int lds_byte_addr, const void* gptr,
                                   unsigned int row_stride_elems,
                                   unsigned int tile_x, unsigned int tile_y) {
#if HAVE_TDM
  unsigned long long ga = (unsigned long long)(uintptr_t)gptr;
  u32x4 g0;
  g0[0] = 1u;                                            // count=1, user D#
  g0[1] = lds_byte_addr;                                 // lds_addr
  g0[2] = (unsigned int)ga;                              // global_addr[31:0]
  g0[3] = (unsigned int)((ga >> 32) & 0x1FFFFFFu) | (2u << 30);  // [56:32]|type=2
  const unsigned int td0 = 0x01000000u;                  // 16M: effectively no OOB
  const unsigned int td1 = 0x01000000u;
  i32x8 g1;
  g1[0] = (int)(1u << 16);                               // data_size = 2 bytes
  g1[1] = (int)((td0 & 0xFFFFu) << 16);                  // tensor_dim0[15:0]
  g1[2] = (int)((td0 >> 16) | ((td1 & 0xFFFFu) << 16));  // td0 hi | td1 lo
  g1[3] = (int)((td1 >> 16) | (tile_x << 16));           // td1 hi | tile_dim0
  g1[4] = (int)(tile_y & 0xFFFFu);                       // tile_dim1 (tile_dim2=0)
  g1[5] = (int)row_stride_elems;                         // dim0_stride[31:0]
  g1[6] = 0;
  g1[7] = 0;
  i32x4 gz = {0, 0, 0, 0};
#if __clang_major__ >= 23
  i32x8 gz8 = {0, 0, 0, 0, 0, 0, 0, 0};
  __builtin_amdgcn_tensor_load_to_lds(g0, g1, gz, gz, gz8, 0);
#else
  __builtin_amdgcn_tensor_load_to_lds(g0, g1, gz, gz, 0);
#endif
#else
  (void)lds_byte_addr; (void)gptr; (void)row_stride_elems;
  (void)tile_x; (void)tile_y;
#endif
}

// ---------------------------------------------------------------------------
// WMMA fragment loaders (16x16x32 bf16, wave32; A and B both use contiguous k = 16*(lane>>4) + e)
// A (16x32, M x K) from row-major LDS [m][ldk]: lane m = lane&15,
//   kh=(lane>>4)*16, elems -> K = { kh..kh+15 } : 2x b128 loads
// B (32x16, K x N):
//   - from transposed LDS [n][32]: two contiguous b128 loads
//   - from natural LDS [k][ldn] (TDM layout): 16 strided u16 loads
// C/D (16x16 f32): lane n = lane&15, row = (lane>>4)*8 + vgpr_index
// ---------------------------------------------------------------------------
__device__ inline v16bf frag_A_ld(const bf16* rowBase, int ldk, int k0, int lane) {
  const bf16* r = rowBase + (size_t)(lane & 15) * ldk + k0 + ((lane >> 4) << 4);
  Frag f;
  f.q[0] = *(const int4*)r;
  f.q[1] = *(const int4*)(r + 8);
  return f.v;
}

__device__ inline v16bf frag_B_t(const bf16 (*tile)[32], int col0, int lane) {
  const bf16* r = tile[col0 + (lane & 15)];
  int kb = (lane >> 4) << 4;
  Frag f;
  f.q[0] = *(const int4*)(r + kb);
  f.q[1] = *(const int4*)(r + kb + 8);
  return f.v;
}

__device__ inline v16bf frag_B_n(const bf16* base, int ldn, int col0, int lane) {
  int n = col0 + (lane & 15);
  int kb = (lane >> 4) << 4;
  v16bf r;
#pragma unroll
  for (int e = 0; e < 16; ++e) r[e] = base[(size_t)(kb + e) * ldn + n];
  return r;
}

__device__ inline v8f wmma_bf16(v16bf a, v16bf b, v8f c) {
  return __builtin_amdgcn_wmma_f32_16x16x32_bf16(false, a, false, b, (short)0, c,
                                                 false, false);
}

__device__ inline float leaky(float x) { return x >= 0.0f ? x : 0.1f * x; }

// ---------------------------------------------------------------------------
// conv_gemm: per-(n,p) GEMM  C[M x 1600] = A_f32[M x 128] * B_bf16[128 x 1600]
// grid = (25 tv-tiles, M/64 m-tiles, NP). block = 256 (8 waves), tile 64x64.
// Full K=128 panel staged once: B via TDM (natural [k][n]), A via threads
// (f32->bf16). Single barrier, then 8 unrolled WMMAs per wave.
// ---------------------------------------------------------------------------
template <bool OUT_BF16, bool HAS_BIAS>
__global__ __launch_bounds__(256)
void conv_gemm(const float* __restrict__ A, int lda,
               const bf16* __restrict__ B,
               const float* __restrict__ bias,
               void* __restrict__ Cout, long cBatch) {
  __shared__ bf16 Bs[128][64];  // [k][n], matches global layout (TDM target)
  __shared__ bf16 As[64][128];  // [m][k]
  const int np  = blockIdx.z;
  const int m0g = blockIdx.y * 64;
  const int n0g = blockIdx.x * 64;
  const int tid = threadIdx.x, lane = tid & 31, wid = tid >> 5;
  const int wm0 = (wid >> 1) * 16, wn0 = (wid & 1) * 32;
  const bf16* Bnp = B + (size_t)np * CC * TV;

#if HAVE_TDM
  if (wid == 0) {
    tdm_load_2d((unsigned int)(uintptr_t)(void*)Bs, Bnp + n0g, TV, 64, 128);
  }
#else
  for (int i = tid; i < (128 * 64) / 8; i += 256) {
    int r = i >> 3;
    int c = (i & 7) * 8;
    *(int4*)&Bs[r][c] = *(const int4*)(Bnp + (size_t)r * TV + n0g + c);
  }
#endif
  // stage A panel with f32->bf16 conversion (overlaps with TDM transfer)
#pragma unroll
  for (int i = tid; i < 64 * 128; i += 256) {
    int m = i >> 7, kk = i & 127;
    As[m][kk] = (bf16)A[(size_t)(m0g + m) * lda + kk];
  }
#if HAVE_TDM
  if (wid == 0) __builtin_amdgcn_s_wait_tensorcnt(0);
#endif
  __syncthreads();

  v8f acc0 = {}, acc1 = {};
#pragma unroll
  for (int ks = 0; ks < 4; ++ks) {
    const int k0 = ks * 32;
    v16bf a  = frag_A_ld(&As[wm0][0], 128, k0, lane);
    v16bf b0 = frag_B_n(&Bs[k0][0], 64, wn0, lane);
    v16bf b1 = frag_B_n(&Bs[k0][0], 64, wn0 + 16, lane);
    acc0 = wmma_bf16(a, b0, acc0);
    acc1 = wmma_bf16(a, b1, acc1);
  }

  const int nc = lane & 15;
  const int mh = (lane >> 4) * 8;
#pragma unroll
  for (int r = 0; r < 8; ++r) {
    int mg = m0g + wm0 + mh + r;
    float bv = HAS_BIAS ? bias[mg] : 0.0f;
    size_t off0 = (size_t)np * (size_t)cBatch + (size_t)mg * TV + n0g;
    float v0 = acc0[r] + bv;
    float v1 = acc1[r] + bv;
    if (OUT_BF16) {
      ((bf16*)Cout)[off0 + wn0 + nc]      = (bf16)v0;
      ((bf16*)Cout)[off0 + wn0 + 16 + nc] = (bf16)v1;
    } else {
      ((float*)Cout)[off0 + wn0 + nc]      = v0;
      ((float*)Cout)[off0 + wn0 + 16 + nc] = v1;
    }
  }
}

// ---------------------------------------------------------------------------
// attn_scores: per (n,s,p): scores[t,q] = sum_{c,v} q[c,t,v]*k[c,q,v] / 800
//              att = att_bias + tanh(scores)*alphat
// qk layout: (NP, 192, TV); q chans = s*32+c, k chans = 96+s*32+c
// ---------------------------------------------------------------------------
__global__ __launch_bounds__(256)
void attn_scores(const bf16* __restrict__ qk, const float* __restrict__ attb,
                 const float* __restrict__ alphat, float* __restrict__ att) {
  __shared__ bf16 Qs[64][32];
  __shared__ bf16 Ks[64][32];  // transposed: [t'][k]
  const int idx = blockIdx.x;  // ((n*S+s)*P+p)
  const int n = idx / (SS * PP);
  const int s = (idx / PP) % SS;
  const int p = idx % PP;
  const int np = n * PP + p;
  const int tid = threadIdx.x, lane = tid & 31, wid = tid >> 5;
  const int wm0 = (wid >> 1) * 16, wn0 = (wid & 1) * 32;
  const size_t baseq = ((size_t)np * 192 + s * CI) * TV;
  const size_t basek = ((size_t)np * 192 + 96 + s * CI) * TV;
  v8f acc0 = {}, acc1 = {};
  for (int k0 = 0; k0 < CI * VV; k0 += 32) {
#pragma unroll
    for (int i = tid; i < 64 * 32; i += 256) {
      int t = i >> 5, kk = i & 31;
      int k = k0 + kk;
      int c = k / VV, v = k - c * VV;
      size_t o = (size_t)c * TV + t * VV + v;
      Qs[t][kk] = qk[baseq + o];
      Ks[t][kk] = qk[basek + o];
    }
    __syncthreads();
    v16bf a  = frag_A_ld(&Qs[wm0][0], 32, 0, lane);
    v16bf b0 = frag_B_t(Ks, wn0, lane);
    v16bf b1 = frag_B_t(Ks, wn0 + 16, lane);
    acc0 = wmma_bf16(a, b0, acc0);
    acc1 = wmma_bf16(a, b1, acc1);
    __syncthreads();
  }
  const float alpha = alphat[s];
  const float inv = 1.0f / (float)(CI * VV);
  const int nc = lane & 15;
  const int mh = (lane >> 4) * 8;
  float* out = att + (size_t)idx * TT * TT;
  const float* ab = attb + (size_t)s * TT * TT;
#pragma unroll
  for (int r = 0; r < 8; ++r) {
    int t  = wm0 + mh + r;
    int q0 = wn0 + nc, q1 = wn0 + 16 + nc;
    out[t * TT + q0] = ab[t * TT + q0] + tanhf(acc0[r] * inv) * alpha;
    out[t * TT + q1] = ab[t * TT + q1] + tanhf(acc1[r] * inv) * alpha;
  }
}

// ---------------------------------------------------------------------------
// attn_apply: per (np, v): y[o,q,v] = sum_s sum_t z_s[o,t,v]*att_s[t,q] + b[o]
// z layout: (NP, S, 128, TV) bf16.  M=128(o), N=64(q), K=64(t) x 3 s-groups.
// ---------------------------------------------------------------------------
__global__ __launch_bounds__(256)
void attn_apply(const bf16* __restrict__ z, const float* __restrict__ att,
                const float* __restrict__ bias, float* __restrict__ y) {
  __shared__ bf16 Zs[128][32];
  __shared__ bf16 Bs[64][32];  // [q][t]
  const int v  = blockIdx.x;
  const int np = blockIdx.y;
  const int tid = threadIdx.x, lane = tid & 31, wid = tid >> 5;
  const int m0 = wid * 16;
  v8f acc[4] = {};
  for (int s = 0; s < SS; ++s) {
    const bf16*  zs = z + ((size_t)np * SS + s) * CC * TV;
    const float* as = att + (((size_t)(np / PP) * SS + s) * PP + (np % PP)) * (TT * TT);
    if (s + 1 < SS)
      __builtin_prefetch(zs + (size_t)CC * TV + (size_t)tid * 64, 0, 0);
    for (int k0 = 0; k0 < TT; k0 += 32) {
#pragma unroll
      for (int i = tid; i < 128 * 32; i += 256) {
        int o = i >> 5, kk = i & 31;
        Zs[o][kk] = zs[(size_t)o * TV + (size_t)(k0 + kk) * VV + v];
      }
#pragma unroll
      for (int i = tid; i < 64 * 32; i += 256) {
        int q = i >> 5, kk = i & 31;
        Bs[q][kk] = (bf16)as[(size_t)(k0 + kk) * TT + q];
      }
      __syncthreads();
      v16bf a = frag_A_ld(&Zs[m0][0], 32, 0, lane);
#pragma unroll
      for (int j = 0; j < 4; ++j) {
        v16bf b = frag_B_t(Bs, j * 16, lane);
        acc[j] = wmma_bf16(a, b, acc[j]);
      }
      __syncthreads();
    }
  }
  const int nc = lane & 15;
  const int mh = (lane >> 4) * 8;
#pragma unroll
  for (int j = 0; j < 4; ++j) {
#pragma unroll
    for (int r = 0; r < 8; ++r) {
      int o = m0 + mh + r;
      int q = j * 16 + nc;
      y[((size_t)np * CC + o) * TV + (size_t)q * VV + v] = acc[j][r] + bias[o];
    }
  }
}

// ---------------------------------------------------------------------------
// Training-mode BN over (N,P,T,V) per channel -> fused scale/shift.
// ---------------------------------------------------------------------------
__global__ __launch_bounds__(256)
void bn_stats(const float* __restrict__ y, const float* __restrict__ gamma,
              const float* __restrict__ beta, float* __restrict__ scale,
              float* __restrict__ shift) {
  const int c = blockIdx.x;
  const int tid = threadIdx.x;
  float s = 0.0f, sq = 0.0f;
  for (int np = 0; np < NP; ++np) {
    const float* row = y + ((size_t)np * CC + c) * TV;
    for (int i = tid; i < TV; i += 256) {
      float val = row[i];
      s += val;
      sq += val * val;
    }
  }
  __shared__ float rs[256], rq[256];
  rs[tid] = s;
  rq[tid] = sq;
  __syncthreads();
  for (int st = 128; st > 0; st >>= 1) {
    if (tid < st) {
      rs[tid] += rs[tid + st];
      rq[tid] += rq[tid + st];
    }
    __syncthreads();
  }
  if (tid == 0) {
    const float cnt = (float)NP * (float)TV;
    float mu  = rs[0] / cnt;
    float var = rq[0] / cnt - mu * mu;
    float sc  = gamma[c] * rsqrtf(var + 1e-5f);
    scale[c] = sc;
    shift[c] = beta[c] - mu * sc;
  }
}

// ---------------------------------------------------------------------------
// Elementwise kernels (grid-stride)
// ---------------------------------------------------------------------------
__global__ void to_bf16_kernel(const float* __restrict__ x, bf16* __restrict__ o) {
  for (size_t i = (size_t)blockIdx.x * blockDim.x + threadIdx.x; i < TOT;
       i += (size_t)gridDim.x * blockDim.x)
    o[i] = (bf16)x[i];
}

__global__ void bn_res_act(const float* __restrict__ inp, const float* __restrict__ y,
                           const float* __restrict__ scale, const float* __restrict__ shift,
                           float* __restrict__ act, bf16* __restrict__ actbf) {
  for (size_t i = (size_t)blockIdx.x * blockDim.x + threadIdx.x; i < TOT;
       i += (size_t)gridDim.x * blockDim.x) {
    int c = (int)((i / TV) % CC);
    float val = leaky(inp[i] + y[i] * scale[c] + shift[c]);
    act[i]   = val;
    actbf[i] = (bf16)val;
  }
}

__global__ void bn_res_out(const float* __restrict__ act, const float* __restrict__ y2,
                           const float* __restrict__ scale, const float* __restrict__ shift,
                           float* __restrict__ out) {
  for (size_t i = (size_t)blockIdx.x * blockDim.x + threadIdx.x; i < TOT;
       i += (size_t)gridDim.x * blockDim.x) {
    int c = (int)((i / TV) % CC);
    out[i] = leaky(act[i] + y2[i] * scale[c] + shift[c]);
  }
}

// ---------------------------------------------------------------------------
// Host launcher
// ---------------------------------------------------------------------------
extern "C" void kernel_launch(void* const* d_in, const int* in_sizes, int n_in,
                              void* d_out, int out_size, void* d_ws, size_t ws_size,
                              hipStream_t stream) {
  (void)in_sizes; (void)n_in; (void)out_size; (void)ws_size;
  const float* x      = (const float*)d_in[0];
  const float* att_t1 = (const float*)d_in[1];
  const float* att_t2 = (const float*)d_in[2];
  const float* alphat = (const float*)d_in[3];
  const float* w_in   = (const float*)d_in[4];
  const float* b_in   = (const float*)d_in[5];
  const float* w_out  = (const float*)d_in[6];
  const float* b_out  = (const float*)d_in[7];
  const float* g_out  = (const float*)d_in[8];
  const float* be_out = (const float*)d_in[9];
  const float* w_ff   = (const float*)d_in[10];
  const float* b_ff   = (const float*)d_in[11];
  const float* g_ff   = (const float*)d_in[12];
  const float* be_ff  = (const float*)d_in[13];

  char* w = (char*)d_ws;
  size_t ofs = 0;
  auto carve = [&](size_t bytes) -> void* {
    void* p = w + ofs;
    ofs += (bytes + 255) & ~(size_t)255;
    return p;
  };
  bf16*  act_bf = (bf16*)carve(TOT * sizeof(bf16));                       //  78.6 MB
  bf16*  big    = (bf16*)carve((size_t)NP * SS * CC * TV * sizeof(bf16)); // 235.9 MB
  float* att    = (float*)carve((size_t)NN * SS * PP * TT * TT * sizeof(float));
  float* y1     = (float*)carve(TOT * sizeof(float));                     // 157.3 MB
  float* act1   = (float*)carve(TOT * sizeof(float));                     // 157.3 MB
  float* scA    = (float*)carve(CC * sizeof(float));
  float* shA    = (float*)carve(CC * sizeof(float));
  float* scB    = (float*)carve(CC * sizeof(float));
  float* shB    = (float*)carve(CC * sizeof(float));

  bf16* qk = big;  // (NP, 192, TV) — lives until attn_scores is done
  bf16* z  = big;  // (NP, S, 128, TV) — written after qk is dead

  auto run_block = [&](const float* inp, const float* attb, float* outp) {
    dim3 blk(256);
    to_bf16_kernel<<<2048, blk, 0, stream>>>(inp, act_bf);
    conv_gemm<true, true><<<dim3(25, 3, NP), blk, 0, stream>>>(
        w_in, CC, act_bf, b_in, (void*)qk, (long)192 * TV);
    attn_scores<<<NN * SS * PP, blk, 0, stream>>>(qk, attb, alphat, att);
    for (int s = 0; s < SS; ++s)
      conv_gemm<true, false><<<dim3(25, 2, NP), blk, 0, stream>>>(
          w_out + s * CC, SS * CC, act_bf, nullptr,
          (void*)(z + (size_t)s * CC * TV), (long)SS * CC * TV);
    attn_apply<<<dim3(VV, NP), blk, 0, stream>>>(z, att, b_out, y1);
    bn_stats<<<CC, blk, 0, stream>>>(y1, g_out, be_out, scA, shA);
    bn_res_act<<<2048, blk, 0, stream>>>(inp, y1, scA, shA, act1, act_bf);
    conv_gemm<false, true><<<dim3(25, 2, NP), blk, 0, stream>>>(
        w_ff, CC, act_bf, b_ff, (void*)y1, (long)CC * TV);
    bn_stats<<<CC, blk, 0, stream>>>(y1, g_ff, be_ff, scB, shB);
    bn_res_out<<<2048, blk, 0, stream>>>(act1, y1, scB, shB, outp);
  };

  run_block(x,             att_t1, (float*)d_out);
  run_block((float*)d_out, att_t2, (float*)d_out);
}
